// ppi_model_got_embs_84344567759044
// MI455X (gfx1250) — matrix operand order repaired and checked by reading, not compiled, not run on hardware
//
#include <hip/hip_runtime.h>
#include <hip/hip_bf16.h>

// ---------------------------------------------------------------------------
// Types for CDNA5 WMMA (gfx1250, wave32)
// ---------------------------------------------------------------------------
typedef __attribute__((ext_vector_type(16))) __bf16          v16bf;
typedef __attribute__((ext_vector_type(16))) unsigned short  v16us;
typedef __attribute__((ext_vector_type(8)))  float           v8f;
typedef __attribute__((ext_vector_type(4)))  unsigned int    v4u;

union FragU { v4u q[2]; v16us us; v16bf bf; };

#define GG      1024
#define NPG0    256
#define NEDGE   2048            // NPG0 * EPG
#define HC      64              // H_GCN
#define HG      128             // H_GIN
#define PPI_E   32768           // G * 32

__device__ inline unsigned short f2bf(float x) {
    union { float f; unsigned int u; } a; a.f = x;
    unsigned int r = a.u + 0x7FFFu + ((a.u >> 16) & 1u);   // round-to-nearest-even
    return (unsigned short)(r >> 16);
}

// CDNA5 async global->LDS copy (ASYNCcnt-tracked), per CDNA5 ISA.
// VDST = LDS byte offset (low 32 bits of the flat shared address), VADDR = 64-bit
// global address (GV mode).
__device__ inline void async_copy_b128(void* lds_ptr, const void* gaddr) {
    unsigned lds_off = (unsigned)(unsigned long long)(uintptr_t)lds_ptr;
    asm volatile("global_load_async_to_lds_b128 %0, %1, off"
                 :: "v"(lds_off), "v"((unsigned long long)(uintptr_t)gaddr)
                 : "memory");
}
__device__ inline void async_wait0() {
    asm volatile("s_wait_asynccnt 0" ::: "memory");
}

// A fragment: 16x32 bf16 tile from LDS row-major [row][64] ushort.
// lane L: m = L&15, half = L>>4; elems 0..7 -> K = kb + half*8 + e,
//                                elems 8..15 -> K = kb + 16 + half*8 + (e-8)
__device__ inline v16bf load_a(const unsigned short* base, int row0, int kb, int lane) {
    int m = lane & 15, half = lane >> 4;
    const unsigned short* r = base + (row0 + m) * 64;
    FragU f;
    f.q[0] = *(const v4u*)(r + kb + half * 8);
    f.q[1] = *(const v4u*)(r + kb + 16 + half * 8);
    return f.bf;
}

// B fragment: 32x16 bf16 tile. W stored transposed in LDS as Wt[nout][64] ushort.
// lane L: n = L&15, half = L>>4; elems 0..7 -> K = kb + half*8 + e, elems 8..15 -> K = kb + 16 + half*8 + (e-8)
__device__ inline v16bf load_b(const unsigned short* base, int ncol0, int kb, int lane) {
    int n = lane & 15, half = lane >> 4;
    const unsigned short* r = base + (ncol0 + n) * 64 + kb + half * 8;
    FragU f;
    f.q[0] = *(const v4u*)(r);
    f.q[1] = *(const v4u*)(r + 16);
    return f.bf;
}

// s_c[n x 64] = s_a[n x 64] @ s_w^T (s_w holds Wt[64][64]); 8 waves cooperate.
__device__ inline void wmma_tile_64(const unsigned short* s_a, const unsigned short* s_w,
                                    float* s_c, int n, int wave, int lane) {
    const int nmt = n >> 4;
    for (int t = wave; t < nmt * 4; t += 8) {      // wave-uniform loop, EXEC stays full
        int mt = t >> 2, nt = t & 3;
        v8f c = {};
#pragma unroll
        for (int kb = 0; kb < 64; kb += 32) {
            v16bf a = load_a(s_a, mt * 16, kb, lane);
            v16bf b = load_b(s_w, nt * 16, kb, lane);
            c = __builtin_amdgcn_wmma_f32_16x16x32_bf16(false, a, false, b,
                                                        (short)0, c, false, false);
        }
        int m = lane & 15, half = lane >> 4;
#pragma unroll
        for (int j = 0; j < 8; ++j)
            s_c[(mt * 16 + j + 8 * half) * 64 + nt * 16 + m] = c[j];
    }
}

// ---------------------------------------------------------------------------
// Stage A: edges -> local ids, ew = 1
// ---------------------------------------------------------------------------
__global__ void k_init_edges(const int* __restrict__ pe, int* srcW, int* dstW,
                             float* ewW, int total) {
    int i = blockIdx.x * 256 + threadIdx.x;
    if (i >= total) return;
    int g = i / NEDGE;
    srcW[i] = pe[i] - g * NPG0;
    dstW[i] = pe[total + i] - g * NPG0;
    ewW[i]  = 1.0f;
}

__global__ void k_zerof(float* p, int n) {
    int i = blockIdx.x * blockDim.x + threadIdx.x;
    if (i < n) p[i] = 0.0f;
}

// ---------------------------------------------------------------------------
// Fused per-graph: GCNConv + FC + ReLU (+ BN partial sums).  1 block == 1 graph.
// Dyn LDS: s_bf(32768) s_w(8192) s_xw(65536) s_out(65536) s_deg(1024)
//          s_esrc(8192) s_edst(8192) s_ew(8192)  = 197632 B
// Edge lists staged global->LDS with async B128 copies, overlapped with conv WMMA.
// ---------------------------------------------------------------------------
__global__ void __launch_bounds__(256)
k_gcn_fc(const float* __restrict__ xin, int feat_in,
         const float* __restrict__ convW, const float* __restrict__ convB,
         const float* __restrict__ fcW,   const float* __restrict__ fcB,
         const int* __restrict__ srcE, const int* __restrict__ dstE,
         const float* __restrict__ ewE,
         float* __restrict__ yout, float* bnSum, float* bnSq, int n) {
    extern __shared__ char smem[];
    unsigned short* s_bf = (unsigned short*)smem;            // 256*64 ushort
    unsigned short* s_w  = (unsigned short*)(smem + 32768);  // 64*64 ushort
    float* s_xw   = (float*)(smem + 40960);                  // 256*64 f32
    float* s_out  = (float*)(smem + 106496);                 // 256*64 f32
    float* s_deg  = (float*)(smem + 172032);                 // 256 f32
    int*   s_esrc = (int*)  (smem + 173056);                 // 2048 i32
    int*   s_edst = (int*)  (smem + 181248);                 // 2048 i32
    float* s_ew   = (float*)(smem + 189440);                 // 2048 f32
    const int g = blockIdx.x, tid = threadIdx.x;
    const int lane = tid & 31, wave = tid >> 5;
    const int nf = n * 64;
    const long ebase = (long)g * NEDGE;

    // ---- issue async edge staging (3 x 8KB), overlapped with conv phase ----
    for (int i = tid; i < NEDGE / 4; i += 256) {             // 512 B128 copies / array
        async_copy_b128(s_esrc + i * 4, srcE + ebase + i * 4);
        async_copy_b128(s_edst + i * 4, dstE + ebase + i * 4);
        async_copy_b128(s_ew   + i * 4, ewE  + ebase + i * 4);
    }

    // ---- conv: xw = x @ convW ----
    if (feat_in == 64) {
        for (int i = tid; i < nf; i += 256) s_bf[i] = f2bf(xin[(long)g * nf + i]);
        for (int i = tid; i < 4096; i += 256) {
            int k = i >> 6, f = i & 63;
            s_w[f * 64 + k] = f2bf(convW[i]);                // transpose to Wt[f][k]
        }
        __syncthreads();
        wmma_tile_64(s_bf, s_w, s_xw, n, wave, lane);
    } else {  // round 1: 7 input features, scalar
        for (int i = tid; i < nf; i += 256) {
            int v = i >> 6, f = i & 63;
            const float* xr = xin + ((long)g * n + v) * 7;
            float acc = 0.0f;
#pragma unroll
            for (int c = 0; c < 7; ++c) acc += xr[c] * convW[c * 64 + f];
            s_xw[i] = acc;
        }
    }
    // ---- degree + symmetric norm (edges now in LDS) ----
    for (int v = tid; v < n; v += 256) s_deg[v] = 1.0f;      // self-loop
    async_wait0();                                           // each wave's copies done
    __syncthreads();                                         // ... visible to all waves
    for (int e = tid; e < NEDGE; e += 256) {
        float w = s_ew[e];
        if (w != 0.0f) atomicAdd(&s_deg[s_edst[e]], w);
    }
    __syncthreads();
    for (int v = tid; v < n; v += 256) s_deg[v] = rsqrtf(s_deg[v]);   // dinv
    __syncthreads();
    for (int i = tid; i < nf; i += 256) {
        int v = i >> 6;
        float di = s_deg[v];
        s_out[i] = di * di * s_xw[i];                        // self-loop term
    }
    __syncthreads();
    for (int e = tid; e < NEDGE; e += 256) {
        float w = s_ew[e];
        if (w != 0.0f) {
            int s = s_esrc[e], d = s_edst[e];
            float coef = w * s_deg[s] * s_deg[d];
            const float* xs = s_xw + s * 64;
            float* od = s_out + d * 64;
#pragma unroll 8
            for (int f = 0; f < 64; ++f) atomicAdd(&od[f], coef * xs[f]);
        }
    }
    __syncthreads();
    // ---- fc: y = relu((out + convB) @ fcW + fcB) ----
    for (int i = tid; i < nf; i += 256) s_bf[i] = f2bf(s_out[i] + convB[i & 63]);
    for (int i = tid; i < 4096; i += 256) {
        int k = i >> 6, f = i & 63;
        s_w[f * 64 + k] = f2bf(fcW[i]);
    }
    __syncthreads();
    wmma_tile_64(s_bf, s_w, s_xw, n, wave, lane);
    __syncthreads();
    float psum = 0.0f, psq = 0.0f;
    for (int i = tid; i < nf; i += 256) {
        float y = fmaxf(s_xw[i] + fcB[i & 63], 0.0f);
        yout[(long)g * nf + i] = y;
        psum += y; psq += y * y;                             // thread's f is fixed (tid&63)
    }
    __syncthreads();
    float* red = (float*)smem;                               // reuse s_bf region
    red[tid] = psum; red[256 + tid] = psq;
    __syncthreads();
    if (tid < 64) {                                          // per-feature block reduce
        float s = red[tid] + red[tid + 64] + red[tid + 128] + red[tid + 192];
        float q = red[256 + tid] + red[320 + tid] + red[384 + tid] + red[448 + tid];
        atomicAdd(&bnSum[tid], s);
        atomicAdd(&bnSq[tid], q);
    }
}

__global__ void k_bn_fin(const float* sum, const float* sq, float* mean, float* rstd,
                         float cinv, int F) {
    int i = blockIdx.x * blockDim.x + threadIdx.x;
    if (i < F) {
        float m = sum[i] * cinv;
        float v = sq[i] * cinv - m * m;
        mean[i] = m;
        rstd[i] = rsqrtf(v + 1e-5f);
    }
}

// ---------------------------------------------------------------------------
// Fused per-graph: BN apply + SAGPool (score, top-k via rank, gather, edge remap)
// Dyn LDS: s_x(65536) s_sc(1024) s_sr(1024) s_ag(1024) s_rk(1024)
//          s_esrc(8192) s_edst(8192) s_ew(8192) = 94208 B
// ---------------------------------------------------------------------------
__global__ void __launch_bounds__(256)
k_sag(const float* __restrict__ y, const float* __restrict__ mean,
      const float* __restrict__ rstd, const float* __restrict__ bng,
      const float* __restrict__ bnb, const float* __restrict__ Wrel,
      const float* __restrict__ brel, const float* __restrict__ Wroot,
      int* srcE, int* dstE, float* ewE, float* __restrict__ xout, int n, int k) {
    extern __shared__ char smem[];
    float* s_x    = (float*)smem;
    float* s_sc   = (float*)(smem + 65536);
    float* s_sr   = (float*)(smem + 66560);
    float* s_ag   = (float*)(smem + 67584);
    int*   s_rk   = (int*)  (smem + 68608);
    int*   s_esrc = (int*)  (smem + 69632);
    int*   s_edst = (int*)  (smem + 77824);
    float* s_ew   = (float*)(smem + 86016);
    const int g = blockIdx.x, tid = threadIdx.x;
    const int nf = n * 64;
    const long ebase = (long)g * NEDGE;

    // ---- async edge staging, overlapped with BN apply + score dots ----
    for (int i = tid; i < NEDGE / 4; i += 256) {
        async_copy_b128(s_esrc + i * 4, srcE + ebase + i * 4);
        async_copy_b128(s_edst + i * 4, dstE + ebase + i * 4);
        async_copy_b128(s_ew   + i * 4, ewE  + ebase + i * 4);
    }

    for (int i = tid; i < nf; i += 256) {
        int f = i & 63;
        s_x[i] = (y[(long)g * nf + i] - mean[f]) * rstd[f] * bng[f] + bnb[f];
    }
    if (tid < n) s_ag[tid] = 0.0f;
    __syncthreads();
    if (tid < n) {
        float rel = 0.0f, root = 0.0f;
        const float* xr = s_x + tid * 64;
#pragma unroll 8
        for (int f = 0; f < 64; ++f) { rel += xr[f] * Wrel[f]; root += xr[f] * Wroot[f]; }
        s_sr[tid] = rel; s_sc[tid] = root;
    }
    async_wait0();
    __syncthreads();
    for (int e = tid; e < NEDGE; e += 256) {
        float w = s_ew[e];
        if (w != 0.0f) atomicAdd(&s_ag[s_edst[e]], w * s_sr[s_esrc[e]]);
    }
    __syncthreads();
    if (tid < n) s_sc[tid] = tanhf(s_ag[tid] + brel[0] + s_sc[tid]);
    __syncthreads();
    int myrank = n;
    if (tid < n) {                       // rank = descending-score order, index tiebreak
        float sv = s_sc[tid];
        int r = 0;
        for (int u = 0; u < n; ++u) {
            float su = s_sc[u];
            r += (su > sv) || (su == sv && u < tid);
        }
        s_rk[tid] = r; myrank = r;
    }
    __syncthreads();
    if (tid < n && myrank < k) {
        float sv = s_sc[tid];
        const float* xr = s_x + tid * 64;
        float* orow = xout + ((long)g * k + myrank) * 64;
#pragma unroll 8
        for (int f = 0; f < 64; ++f) orow[f] = xr[f] * sv;
    }
    for (int e = tid; e < NEDGE; e += 256) {
        int s = s_esrc[e], d = s_edst[e];
        float w = s_ew[e];
        int rs = s_rk[s], rd = s_rk[d];
        bool ks = rs < k, kd = rd < k;
        ewE[ebase + e]  = (ks && kd) ? w : 0.0f;
        srcE[ebase + e] = ks ? rs : 0;
        dstE[ebase + e] = kd ? rd : 0;
    }
}

__global__ void k_meanpool(const float* __restrict__ x, float* __restrict__ embs, int n) {
    int i = blockIdx.x * 256 + threadIdx.x;
    if (i >= GG * 64) return;
    int g = i >> 6, f = i & 63;
    float s = 0.0f;
    for (int r = 0; r < n; ++r) s += x[((long)g * n + r) * 64 + f];
    embs[i] = s * (1.0f / (float)n);
}

// ---------------------------------------------------------------------------
// Stage B: GIN over protein graph
// ---------------------------------------------------------------------------
__global__ void k_scale(const float* __restrict__ h, const float* __restrict__ eps,
                        float* __restrict__ z, int total) {
    int i = blockIdx.x * 256 + threadIdx.x;
    if (i < total) z[i] = (1.0f + eps[0]) * h[i];
}

__global__ void k_edge_add(const float* __restrict__ h, const int* __restrict__ ei,
                           float* z, int F) {
    int i = blockIdx.x * 256 + threadIdx.x;
    int total = PPI_E * F;
    if (i >= total) return;
    int e = i / F, f = i - e * F;
    int s = ei[e], d = ei[PPI_E + e];
    atomicAdd(&z[d * F + f], h[s * F + f]);
}

// Generic WMMA GEMM: C[MxN] = op(A[MxK] @ W[KxN] + bias); optional BN stat accum.
__global__ void __launch_bounds__(256)
k_gemm_bf16(const float* __restrict__ A, const float* __restrict__ W,
            const float* __restrict__ bias, float* __restrict__ C,
            int M, int K, int N, int relu, float* bnSum, float* bnSq) {
    int wave = threadIdx.x >> 5, lane = threadIdx.x & 31;
    int ntiles = N >> 4;
    int tiles = (M >> 4) * ntiles;
    int t = blockIdx.x * 8 + wave;
    if (t >= tiles) return;                       // wave-uniform
    int mt = t / ntiles, nt = t - mt * ntiles;
    int m = lane & 15, half = lane >> 4;
    v8f c = {};
    for (int kb = 0; kb < K; kb += 32) {
        FragU fa, fb;
        const float* arow = A + (long)(mt * 16 + m) * K + kb;
#pragma unroll
        for (int e = 0; e < 8; ++e) fa.us[e] = f2bf(arow[half * 8 + e]);
#pragma unroll
        for (int e = 0; e < 8; ++e) fa.us[8 + e] = f2bf(arow[16 + half * 8 + e]);
        const float* wcol = W + nt * 16 + m;
#pragma unroll
        for (int e = 0; e < 16; ++e) fb.us[e] = f2bf(wcol[(long)(kb + (e >> 3) * 16 + half * 8 + (e & 7)) * N]);
        c = __builtin_amdgcn_wmma_f32_16x16x32_bf16(false, fa.bf, false, fb.bf,
                                                    (short)0, c, false, false);
    }
    int col = nt * 16 + m;
    float bs = bias ? bias[col] : 0.0f;
    float psum = 0.0f, psq = 0.0f;
#pragma unroll
    for (int j = 0; j < 8; ++j) {
        float v = c[j] + bs;
        if (relu) v = fmaxf(v, 0.0f);
        C[(long)(mt * 16 + j + 8 * half) * N + col] = v;
        psum += v; psq += v * v;
    }
    if (bnSum) { atomicAdd(&bnSum[col], psum); atomicAdd(&bnSq[col], psq); }
}

__global__ void k_bn_apply(const float* __restrict__ u, const float* __restrict__ mean,
                           const float* __restrict__ rstd, const float* __restrict__ gg,
                           const float* __restrict__ bb, float* __restrict__ h,
                           int F, int total) {
    int i = blockIdx.x * 256 + threadIdx.x;
    if (i >= total) return;
    int f = i % F;
    h[i] = (u[i] - mean[f]) * rstd[f] * gg[f] + bb[f];
}

__global__ void k_final(const float* __restrict__ h, const float* __restrict__ w,
                        const float* __restrict__ b, float* __restrict__ out) {
    int r = blockIdx.x * 256 + threadIdx.x;
    if (r >= GG) return;
    float acc = b[0];
    const float* hr = h + (long)r * HG;
#pragma unroll 8
    for (int f = 0; f < HG; ++f) acc += hr[f] * w[f];
    out[r] = acc;
}

// ---------------------------------------------------------------------------
// Host orchestration
// ---------------------------------------------------------------------------
// params order: 0 batch, 1 p_x_all, 2 edge_index, 3 p_edge_all, then params dict
// in insertion order.
enum {
    IN_BATCH = 0, IN_PX, IN_EI, IN_PE,
    CONV1_W = 4, CONV1_B, CONV2_W, CONV2_B, CONV3_W, CONV3_B, CONV4_W, CONV4_B,
    FC1_W = 12,                           // each round block: fcW,fcB,bnG,bnB,Wrel,brel,Wroot
    EPS1 = 40, EPS2 = 41, EPS3 = 42,
    G1_W1 = 43, G1_B1, G1_W2, G1_B2, G1_BNG, G1_BNB,
    G2_W1 = 49, G2_B1, G2_BNG, G2_BNB,
    G3_W1 = 53, G3_B1, G3_W2, G3_B2, G3_BNG, G3_BNB,
    LIN1_W = 59, LIN1_B = 60, FCO_W = 61, FCO_B = 62
};

extern "C" void kernel_launch(void* const* d_in, const int* in_sizes, int n_in,
                              void* d_out, int out_size, void* d_ws, size_t ws_size,
                              hipStream_t stream) {
    (void)in_sizes; (void)n_in; (void)out_size; (void)ws_size;
    auto F32 = [&](int i) { return (const float*)d_in[i]; };
    auto I32 = [&](int i) { return (const int*)d_in[i]; };

    char* ws = (char*)d_ws;
    float* bufX   = (float*)(ws);                  // 1024*256*64 f32 = 64 MB
    float* bufY   = (float*)(ws + 67108864);       // 64 MB
    int*   srcW   = (int*)  (ws + 134217728);      // 8 MB
    int*   dstW   = (int*)  (ws + 142606336);      // 8 MB
    float* ewW    = (float*)(ws + 150994944);      // 8 MB
    float* bnSum  = (float*)(ws + 159383552);
    float* bnSq   = (float*)(ws + 159384064);
    float* bnMean = (float*)(ws + 159384576);
    float* bnRstd = (float*)(ws + 159385088);
    float* zbuf   = (float*)(ws + 159385600);      // 1024*128
    float* ubuf   = (float*)(ws + 159909888);
    float* tbuf   = (float*)(ws + 160434176);
    float* hbuf   = (float*)(ws + 160958464);

    float* out  = (float*)d_out;       // final[1024] then embs[1024*64]
    float* embs = out + GG;

    const size_t LDS_GCN = 197632, LDS_SAG = 94208;
    const int totE = GG * NEDGE;

    k_init_edges<<<(totE + 255) / 256, 256, 0, stream>>>(I32(IN_PE), srcW, dstW, ewW, totE);

    // ---- Stage A: 4 fused rounds ----
    const float* xin = F32(IN_PX);
    int feat = 7, n = NPG0;
    const int convWi[4] = {CONV1_W, CONV2_W, CONV3_W, CONV4_W};
    for (int r = 0; r < 4; ++r) {
        int base = FC1_W + r * 7;     // fcW, fcB, bnG, bnB, Wrel, brel, Wroot
        int k = n >> 1;
        k_zerof<<<1, 128, 0, stream>>>(bnSum, 64);
        k_zerof<<<1, 128, 0, stream>>>(bnSq, 64);
        k_gcn_fc<<<GG, 256, LDS_GCN, stream>>>(
            xin, feat, F32(convWi[r]), F32(convWi[r] + 1),
            F32(base + 0), F32(base + 1), srcW, dstW, ewW, bufY, bnSum, bnSq, n);
        k_bn_fin<<<1, 64, 0, stream>>>(bnSum, bnSq, bnMean, bnRstd,
                                       1.0f / ((float)GG * (float)n), 64);
        k_sag<<<GG, 256, LDS_SAG, stream>>>(
            bufY, bnMean, bnRstd, F32(base + 2), F32(base + 3),
            F32(base + 4), F32(base + 5), F32(base + 6),
            srcW, dstW, ewW, bufX, n, k);
        xin = bufX; feat = 64; n = k;
    }
    k_meanpool<<<(GG * 64 + 255) / 256, 256, 0, stream>>>(bufX, embs, 16);   // n == 16

    // ---- Stage B: GIN ----
    auto gemm = [&](const float* A, const float* W, const float* B, float* C,
                    int M, int K, int N, int relu, float* s, float* q) {
        int tiles = (M / 16) * (N / 16);
        k_gemm_bf16<<<(tiles + 7) / 8, 256, 0, stream>>>(A, W, B, C, M, K, N, relu, s, q);
    };
    // layer 1: z = (1+eps1)*embs + agg ; t = relu(z@W1+b1); u = relu(t@W2+b2); h = BN(u)
    k_scale<<<(GG * 64 + 255) / 256, 256, 0, stream>>>(embs, F32(EPS1), zbuf, GG * 64);
    k_edge_add<<<(PPI_E * 64 + 255) / 256, 256, 0, stream>>>(embs, I32(IN_EI), zbuf, 64);
    gemm(zbuf, F32(G1_W1), F32(G1_B1), tbuf, GG, 64, 128, 1, nullptr, nullptr);
    k_zerof<<<1, 128, 0, stream>>>(bnSum, 128);
    k_zerof<<<1, 128, 0, stream>>>(bnSq, 128);
    gemm(tbuf, F32(G1_W2), F32(G1_B2), ubuf, GG, 128, 128, 1, bnSum, bnSq);
    k_bn_fin<<<1, 128, 0, stream>>>(bnSum, bnSq, bnMean, bnRstd, 1.0f / (float)GG, 128);
    k_bn_apply<<<(GG * 128 + 255) / 256, 256, 0, stream>>>(
        ubuf, bnMean, bnRstd, F32(G1_BNG), F32(G1_BNB), hbuf, 128, GG * 128);
    // layer 2
    k_scale<<<(GG * 128 + 255) / 256, 256, 0, stream>>>(hbuf, F32(EPS2), zbuf, GG * 128);
    k_edge_add<<<(PPI_E * 128 + 255) / 256, 256, 0, stream>>>(hbuf, I32(IN_EI), zbuf, 128);
    k_zerof<<<1, 128, 0, stream>>>(bnSum, 128);
    k_zerof<<<1, 128, 0, stream>>>(bnSq, 128);
    gemm(zbuf, F32(G2_W1), F32(G2_B1), ubuf, GG, 128, 128, 1, bnSum, bnSq);
    k_bn_fin<<<1, 128, 0, stream>>>(bnSum, bnSq, bnMean, bnRstd, 1.0f / (float)GG, 128);
    k_bn_apply<<<(GG * 128 + 255) / 256, 256, 0, stream>>>(
        ubuf, bnMean, bnRstd, F32(G2_BNG), F32(G2_BNB), hbuf, 128, GG * 128);
    // layer 3
    k_scale<<<(GG * 128 + 255) / 256, 256, 0, stream>>>(hbuf, F32(EPS3), zbuf, GG * 128);
    k_edge_add<<<(PPI_E * 128 + 255) / 256, 256, 0, stream>>>(hbuf, I32(IN_EI), zbuf, 128);
    gemm(zbuf, F32(G3_W1), F32(G3_B1), tbuf, GG, 128, 128, 1, nullptr, nullptr);
    k_zerof<<<1, 128, 0, stream>>>(bnSum, 128);
    k_zerof<<<1, 128, 0, stream>>>(bnSq, 128);
    gemm(tbuf, F32(G3_W2), F32(G3_B2), ubuf, GG, 128, 128, 1, bnSum, bnSq);
    k_bn_fin<<<1, 128, 0, stream>>>(bnSum, bnSq, bnMean, bnRstd, 1.0f / (float)GG, 128);
    k_bn_apply<<<(GG * 128 + 255) / 256, 256, 0, stream>>>(
        ubuf, bnMean, bnRstd, F32(G3_BNG), F32(G3_BNB), hbuf, 128, GG * 128);
    // lin1 + final head
    gemm(hbuf, F32(LIN1_W), F32(LIN1_B), tbuf, GG, 128, 128, 1, nullptr, nullptr);
    k_final<<<(GG + 255) / 256, 256, 0, stream>>>(tbuf, F32(FCO_W), F32(FCO_B), out);
}
